// SRHDR_59811714564189
// MI455X (gfx1250) — compile-verified, matrix operand order checked by reading, not run on hardware
//
#include <hip/hip_runtime.h>
#include <hip/hip_bf16.h>

#define DEV static __device__ __forceinline__

typedef __attribute__((ext_vector_type(16))) _Float16 v16h;
typedef __attribute__((ext_vector_type(8)))  float    v8f;

// ---------------- constants ----------------
// dims: B=1, E=4, C=32, H=W=192, WS=8, N=64, NH=4, nh=nw=24, nWin=576
// windows total = E*nWin = 2304 ; groups Bn = 576 ; HW = 36864
constexpr float  SCALE   = 0.17677669529663687f;  // 32^-0.5
constexpr int    HW      = 36864;                 // 192*192

// f16 weight arena offsets (in halfs)
constexpr size_t OFF_WEA  = 0;                    // 32x384
constexpr size_t OFF_WSA  = 12288;                // 32x384
constexpr size_t OFF_FC11 = 24576;                // 1536x128
constexpr size_t OFF_FC12 = 221184;               // 1536x128
constexpr size_t OFF_FC2  = 417792;               // 128x128
constexpr size_t OFF_M11  = 434176;               // 128x128
constexpr size_t OFF_M12  = 450560;               // 128x128
constexpr size_t OFF_M2   = 466944;               // 128x128
constexpr size_t W16_TOT  = 483328;

// ---------------- WMMA helpers (ISA 7.12.2 layouts) ----------------
DEV int wmma_k(int j, int hi) { return j + (j & 8) + (hi << 3); }  // K index map

DEV v16h ldA(const _Float16* p, int ld) {        // A: 16x32 row-major
  int lane = threadIdx.x & 31;
  int m = lane & 15, hi = lane >> 4;
  v16h a;
#pragma unroll
  for (int j = 0; j < 16; ++j) a[j] = p[m * ld + wmma_k(j, hi)];
  return a;
}

DEV v16h ldA_scaled(const _Float16* p, int ld, float s) {
  int lane = threadIdx.x & 31;
  int m = lane & 15, hi = lane >> 4;
  v16h a;
#pragma unroll
  for (int j = 0; j < 16; ++j)
    a[j] = (_Float16)((float)p[m * ld + wmma_k(j, hi)] * s);
  return a;
}

DEV v16h ldB(const _Float16* p, int ld) {        // B: 32x16 row-major (KxN)
  int lane = threadIdx.x & 31;
  int n = lane & 15, hi = lane >> 4;
  v16h b;
#pragma unroll
  for (int j = 0; j < 16; ++j) b[j] = p[wmma_k(j, hi) * ld + n];
  return b;
}

DEV v8f wmma(v16h a, v16h b, v8f c) {
  return __builtin_amdgcn_wmma_f32_16x16x32_f16(false, a, false, b, (short)0, c,
                                                false, false);
}

DEV void stC_f16(_Float16* p, int ld, v8f c) {
  int lane = threadIdx.x & 31;
  int n = lane & 15, hi = lane >> 4;
#pragma unroll
  for (int r = 0; r < 8; ++r) p[(r + hi * 8) * ld + n] = (_Float16)c[r];
}

DEV void stC_f32(float* p, int ld, v8f c) {
  int lane = threadIdx.x & 31;
  int n = lane & 15, hi = lane >> 4;
#pragma unroll
  for (int r = 0; r < 8; ++r) p[(r + hi * 8) * ld + n] = c[r];
}

// Low 32 bits of a generic pointer into LDS == LDS byte offset (ISA §10.2:
// LDS aperture maps addr[31:0] to the wave's LDS allocation).
DEV unsigned ldsOff(const void* p) { return (unsigned)(size_t)p; }

// Async copy of one 128-bit packet per lane: LDS[dst] = MEM[sbase + voff].
// GVS addressing; tracked by ASYNCcnt (ISA §10 / 15.18.3 op 98).
DEV void async_g2l_b128(unsigned lds_dst, unsigned voff, const void* sbase) {
  unsigned long long sb = (unsigned long long)sbase;
  asm volatile("global_load_async_to_lds_b128 %0, %1, %2"
               :: "v"(lds_dst), "v"(voff), "s"(sb)
               : "memory");
}

DEV void wait_async0() {
  asm volatile("s_wait_asynccnt 0x0" ::: "memory");
}

// ---------------- kernels ----------------

// f32 -> f16 weight conversion
__global__ void k_cvt(const float* __restrict__ s, _Float16* __restrict__ d, int n) {
  int i = blockIdx.x * 256 + threadIdx.x;
  if (i < n) d[i] = (_Float16)s[i];
}

// window partition (faithful scrambled view) + LN1 + sine pos-enc
// widx = e*576 + hb*24 + wb ; thread = scrambled token n (0..63)
// xw[widx][n][c] = x[e, n>>1, hb*8 + (n&1)*4 + (c>>3), wb*8 + (c&7)]
__global__ void __launch_bounds__(64)
k_prep(const float* __restrict__ x, const float* __restrict__ g,
       const float* __restrict__ bt, _Float16* __restrict__ xw_sa,
       _Float16* __restrict__ xw_ea) {
  int widx = blockIdx.x;
  int n = threadIdx.x;
  int e = widx / 576, win = widx % 576;
  int hb = win / 24, wb = win % 24;
  int corig = n >> 1;
  int ibase = (n & 1) * 4;
  const float* px = x + ((size_t)e * 32 + corig) * HW;
  float v[32];
  float mean = 0.f;
#pragma unroll
  for (int c = 0; c < 32; ++c) {
    int i = ibase + (c >> 3), j = c & 7;
    float val = px[(hb * 8 + i) * 192 + (wb * 8 + j)];
    v[c] = val; mean += val;
  }
  mean *= (1.f / 32.f);
  float var = 0.f;
#pragma unroll
  for (int c = 0; c < 32; ++c) { float d = v[c] - mean; var += d * d; }
  float inv = rsqrtf(var * (1.f / 32.f) + 1e-5f);
  // DETR sine PE in the scrambled token space: row = n>>3, col = n&7
  float ye = (float)((n >> 3) + 1) * (6.283185307179586f / (8.f + 1e-6f));
  float xe = (float)((n & 7) + 1) * (6.283185307179586f / (8.f + 1e-6f));
  size_t base = ((size_t)widx * 64 + n) * 32;
#pragma unroll
  for (int c = 0; c < 32; ++c) {
    float ln = (v[c] - mean) * inv * g[c] + bt[c];
    xw_sa[base + c] = (_Float16)ln;
    int cc = c & 15;
    float emb = (c < 16) ? ye : xe;
    float tk  = __powf(10000.f, (float)(cc >> 1) * 0.125f);
    float ang = emb / tk;
    float pe  = ((cc & 1) == 0) ? __sinf(ang) : __cosf(ang);
    xw_ea[base + c] = (_Float16)(ln + pe);
  }
}

// QKV projections: y = A(64x32) @ W(32x384) for ea (A=xw_ea) and sa (A=xw_sa)
// grid (2304, 48): y-tile 0..23 -> ea, 24..47 -> sa ; one wave per block
__global__ void __launch_bounds__(32)
k_qkv(const _Float16* __restrict__ xw_ea, const _Float16* __restrict__ xw_sa,
      const _Float16* __restrict__ w16, _Float16* __restrict__ y_ea,
      _Float16* __restrict__ y_sa) {
  int widx = blockIdx.x;
  int nt = blockIdx.y;
  bool sa = nt >= 24;
  int ntl = sa ? nt - 24 : nt;
  const _Float16* A = (sa ? xw_sa : xw_ea) + (size_t)widx * 2048;
  const _Float16* W = w16 + (sa ? OFF_WSA : OFF_WEA) + ntl * 16;
  _Float16* Y = (sa ? y_sa : y_ea) + (size_t)widx * 64 * 384 + ntl * 16;
  v16h bm = ldB(W, 384);
#pragma unroll
  for (int mt = 0; mt < 4; ++mt) {
    v16h am = ldA(A + mt * 16 * 32, 32);
    v8f c = {};
    c = wmma(am, bm, c);
    stC_f16(Y + mt * 16 * 384, 384, c);
  }
}

// Exposure attention: grid (576, 4, 8) z = dir*4 + e ; 1 wave per block
// dir0 (ea1): attn(q0, k_e, v_e) ; dir1 (ea2): attn(q_e, k0, v0)
__global__ void __launch_bounds__(32)
k_ea(const _Float16* __restrict__ y_ea, _Float16* __restrict__ feat) {
  __shared__ float    S[64 * 64];
  __shared__ _Float16 P[64 * 64];
  int g = blockIdx.x, h = blockIdx.y;
  int dir = blockIdx.z >> 2, e = blockIdx.z & 3;
  const _Float16* b0 = y_ea + (size_t)g * 64 * 384;               // e=0 window
  const _Float16* be = y_ea + (size_t)(e * 576 + g) * 64 * 384;   // e window
  const _Float16* Q = (dir == 0 ? b0 : be) + h * 32;              // p=0
  const _Float16* K = (dir == 0 ? be : b0) + 128 + h * 32;        // p=1
  const _Float16* V = (dir == 0 ? be : b0) + 256 + h * 32;        // p=2
  // S = (Q*scale) K^T : K=32 -> one WMMA per 16x16 tile
  for (int mt = 0; mt < 4; ++mt) {
    v16h a = ldA_scaled(Q + mt * 16 * 384, 384, SCALE);
    for (int nt = 0; nt < 4; ++nt) {
      v16h bT = ldA(K + nt * 16 * 384, 384);    // K^T tile == A-pattern on K
      v8f c = {};
      c = wmma(a, bT, c);
      stC_f32(S + mt * 16 * 64 + nt * 16, 64, c);
    }
  }
  __syncthreads();
  // softmax: 2 rows per lane
  int lane = threadIdx.x;
#pragma unroll
  for (int rr = 0; rr < 2; ++rr) {
    int row = lane * 2 + rr;
    float* sr = S + row * 64;
    float mx = -1e30f;
    for (int m = 0; m < 64; ++m) mx = fmaxf(mx, sr[m]);
    float sum = 0.f;
    for (int m = 0; m < 64; ++m) { float ex = __expf(sr[m] - mx); sr[m] = ex; sum += ex; }
    float rc = 1.f / sum;
    for (int m = 0; m < 64; ++m) P[row * 64 + m] = (_Float16)(sr[m] * rc);
  }
  __syncthreads();
  // O = P @ V -> feat cols [512*(1+dir) + h*128 + e*32 ...)
  _Float16* out = feat + (size_t)g * 64 * 1536 + 512 * (1 + dir) + h * 128 + e * 32;
  for (int mt = 0; mt < 4; ++mt)
    for (int nt = 0; nt < 2; ++nt) {
      v8f c = {};
#pragma unroll
      for (int ks = 0; ks < 2; ++ks) {
        v16h a  = ldA(P + mt * 16 * 64 + ks * 32, 64);
        v16h bm = ldB(V + ks * 32 * 384 + nt * 16, 384);
        c = wmma(a, bm, c);
      }
      stC_f16(out + mt * 16 * 1536 + nt * 16, 1536, c);
    }
}

// Self-attention (head dim 128 = 4 exposures x 32) with rel-pos bias.
// grid (576, 4), 128 threads (4 waves); wave w owns row-tile mt=w.
// K-dim 128 split into 4 K-steps, each reading the e=ks exposure's window.
__global__ void __launch_bounds__(128)
k_sa(const _Float16* __restrict__ y_sa, const float* __restrict__ rpb,
     _Float16* __restrict__ feat) {
  __shared__ float    S[64 * 64];
  __shared__ _Float16 P[64 * 64];
  int g = blockIdx.x, h = blockIdx.y;
  int tid = threadIdx.x, wave = tid >> 5, lane = tid & 31;
  int m16 = lane & 15, hi = lane >> 4;
  size_t wbase[4];
#pragma unroll
  for (int e = 0; e < 4; ++e) wbase[e] = (size_t)(e * 576 + g) * 64 * 384;
  int mt = wave;
  for (int nt = 0; nt < 4; ++nt) {
    v8f c = {};
#pragma unroll
    for (int ks = 0; ks < 4; ++ks) {
      v16h a, bT;
#pragma unroll
      for (int j = 0; j < 16; ++j) {
        int kk = wmma_k(j, hi);
        a[j]  = (_Float16)((float)y_sa[wbase[ks] + (size_t)(mt * 16 + m16) * 384 +
                                       h * 32 + kk] * SCALE);
        bT[j] = y_sa[wbase[ks] + (size_t)(nt * 16 + m16) * 384 + 128 + h * 32 + kk];
      }
      c = wmma(a, bT, c);
    }
    stC_f32(S + mt * 16 * 64 + nt * 16, 64, c);
  }
  __syncthreads();
  // rel-pos bias + softmax: one row per thread (threads 0..63)
  if (tid < 64) {
    int row = tid, i1 = row >> 3, j1 = row & 7;
    float* sr = S + row * 64;
    float mx = -1e30f;
    for (int m = 0; m < 64; ++m) {
      int i2 = m >> 3, j2 = m & 7;
      float vv = sr[m] + rpb[((i1 - i2 + 7) * 15 + (j1 - j2 + 7)) * 4 + h];
      sr[m] = vv;
      mx = fmaxf(mx, vv);
    }
    float sum = 0.f;
    for (int m = 0; m < 64; ++m) { float ex = __expf(sr[m] - mx); sr[m] = ex; sum += ex; }
    float rc = 1.f / sum;
    for (int m = 0; m < 64; ++m) P[row * 64 + m] = (_Float16)(sr[m] * rc);
  }
  __syncthreads();
  // O = P(64x64) @ V(64x128) -> feat cols [h*128 ...)
  _Float16* out = feat + (size_t)g * 64 * 1536 + h * 128;
  for (int nt = 0; nt < 8; ++nt) {
    int e = nt >> 1, c0 = (nt & 1) * 16;
    v8f c = {};
#pragma unroll
    for (int ks = 0; ks < 2; ++ks) {
      v16h a = ldA(P + mt * 16 * 64 + ks * 32, 64);
      v16h bm;
#pragma unroll
      for (int j = 0; j < 16; ++j) {
        int kk = wmma_k(j, hi);
        bm[j] = y_sa[wbase[e] + (size_t)(ks * 32 + kk) * 384 + 256 + h * 32 + c0 + m16];
      }
      c = wmma(a, bm, c);
    }
    stC_f16(out + mt * 16 * 1536 + nt * 16, 1536, c);
  }
}

// GEGLU #1: feat(64x1536) @ fc11|fc12 -> gate -> @ fc2 -> un-window + residual
// grid 576, 256 threads (8 waves). K streamed in 12 chunks of 128 through LDS
// using async global->LDS b128 copies (ASYNCcnt) + weight prefetch.
__global__ void __launch_bounds__(256)
k_geglu1(const _Float16* __restrict__ feat, const _Float16* __restrict__ w16,
         const float* __restrict__ b11, const float* __restrict__ b12,
         const float* __restrict__ b2, const float* __restrict__ x,
         float* __restrict__ x1) {
  __shared__ _Float16 Ach[64 * 128];
  __shared__ _Float16 Hh[64 * 256];
  int g = blockIdx.x, tid = threadIdx.x, wave = tid >> 5;
  const _Float16* W11 = w16 + OFF_FC11;
  const _Float16* W12 = w16 + OFF_FC12;
  const _Float16* fbase = feat + (size_t)g * 64 * 1536;  // this window's rows
  unsigned lds0 = ldsOff(Ach);
  v8f zero = {};
  v8f acc[8];
#pragma unroll
  for (int t = 0; t < 8; ++t) acc[t] = zero;
  for (int kc = 0; kc < 12; ++kc) {
    __syncthreads();  // all waves done reading previous chunk
    // async-stage 64x128 f16 chunk: 4 b128 packets per lane
#pragma unroll
    for (int it = 0; it < 4; ++it) {
      int idx = (tid + it * 256) * 8;        // half-index, 8 halfs per packet
      int row = idx >> 7, colh = idx & 127;
      unsigned voff = (unsigned)(row * 3072 + colh * 2);   // bytes into window
      async_g2l_b128(lds0 + (unsigned)idx * 2, voff,
                     (const void*)(fbase + kc * 128));
    }
    // prefetch next weight chunk while the copy + WMMAs run
    if (kc + 1 < 12) {
      __builtin_prefetch(W11 + (size_t)(kc + 1) * 128 * 128 + tid * 64, 0, 3);
      __builtin_prefetch(W12 + (size_t)(kc + 1) * 128 * 128 + tid * 64, 0, 3);
    }
    wait_async0();
    __syncthreads();
    for (int mt = 0; mt < 4; ++mt)
      for (int tsub = 0; tsub < 2; ++tsub) {
        int ntt = wave * 2 + tsub;
        const _Float16* W = (ntt < 8) ? W11 : W12;
        int n0 = (ntt & 7) * 16;
        int ti = mt * 2 + tsub;
#pragma unroll
        for (int ks = 0; ks < 4; ++ks) {
          v16h a  = ldA(Ach + mt * 16 * 128 + ks * 32, 128);
          v16h bm = ldB(W + (size_t)(kc * 128 + ks * 32) * 128 + n0, 128);
          acc[ti] = wmma(a, bm, acc[ti]);
        }
      }
  }
  for (int mt = 0; mt < 4; ++mt)
    for (int tsub = 0; tsub < 2; ++tsub) {
      int ntt = wave * 2 + tsub;
      stC_f16(Hh + mt * 16 * 256 + ntt * 16, 256, acc[mt * 2 + tsub]);
    }
  __syncthreads();
  // gate: gelu(h1+b11) * (h2+b12) -> Ach (f16)
  for (int idx = tid; idx < 8192; idx += 256) {
    int r = idx >> 7, jc = idx & 127;
    float h1 = (float)Hh[r * 256 + jc] + b11[jc];
    float h2 = (float)Hh[r * 256 + 128 + jc] + b12[jc];
    float ge = 0.5f * h1 * (1.f + erff(h1 * 0.7071067811865475f));
    Ach[idx] = (_Float16)(ge * h2);
  }
  __syncthreads();
  // out = G @ fc2 + b2, then faithful un-window scatter + residual add
  const _Float16* W2 = w16 + OFF_FC2;
  int hb = g / 24, wb = g % 24;
  int lane = tid & 31, nloc = lane & 15, hi = lane >> 4;
  for (int tt = 0; tt < 4; ++tt) {
    int ti = wave * 4 + tt, mt = ti >> 3, nt = ti & 7;
    v8f c = {};
#pragma unroll
    for (int ks = 0; ks < 4; ++ks) {
      v16h a  = ldA(Ach + mt * 16 * 128 + ks * 32, 128);
      v16h bm = ldB(W2 + ks * 32 * 128 + nt * 16, 128);
      c = wmma(a, bm, c);
    }
    int d = nt * 16 + nloc;
    int e = d >> 5, ch = d & 31;
#pragma unroll
    for (int r = 0; r < 8; ++r) {
      int row = mt * 16 + hi * 8 + r;
      int i = row >> 3, j = row & 7;
      size_t off = ((size_t)e * 32 + ch) * HW + (size_t)(i * 24 + hb) * 192 +
                   (j * 24 + wb);
      x1[off] = x[off] + c[r] + b2[d];
    }
  }
}

// LN2 + GEGLU MLP2 + final residual. grid 576, 256 threads; 64 pixels/block.
__global__ void __launch_bounds__(256)
k_mlp2(const float* __restrict__ x1, const float* __restrict__ g2,
       const float* __restrict__ bt2, const _Float16* __restrict__ w16,
       const float* __restrict__ mb11, const float* __restrict__ mb12,
       const float* __restrict__ mb2, float* __restrict__ out) {
  __shared__ _Float16 A[64 * 128];
  __shared__ _Float16 Hh[64 * 256];
  int tid = threadIdx.x, wave = tid >> 5;
  int pix0 = blockIdx.x * 64;
  if (tid < 64) {  // LN over 128 channels for one pixel per thread
    float v[128];
    float mean = 0.f;
    for (int d = 0; d < 128; ++d) {
      v[d] = x1[(size_t)d * HW + pix0 + tid];
      mean += v[d];
    }
    mean *= (1.f / 128.f);
    float var = 0.f;
    for (int d = 0; d < 128; ++d) { float dv = v[d] - mean; var += dv * dv; }
    float inv = rsqrtf(var * (1.f / 128.f) + 1e-5f);
    for (int d = 0; d < 128; ++d)
      A[tid * 128 + d] = (_Float16)((v[d] - mean) * inv * g2[d] + bt2[d]);
  }
  __syncthreads();
  const _Float16* W11 = w16 + OFF_M11;
  const _Float16* W12 = w16 + OFF_M12;
  for (int mt = 0; mt < 4; ++mt)
    for (int tsub = 0; tsub < 2; ++tsub) {
      int ntt = wave * 2 + tsub;
      const _Float16* W = (ntt < 8) ? W11 : W12;
      int n0 = (ntt & 7) * 16;
      v8f c = {};
#pragma unroll
      for (int ks = 0; ks < 4; ++ks) {
        v16h a  = ldA(A + mt * 16 * 128 + ks * 32, 128);
        v16h bm = ldB(W + ks * 32 * 128 + n0, 128);
        c = wmma(a, bm, c);
      }
      stC_f16(Hh + mt * 16 * 256 + ntt * 16, 256, c);
    }
  __syncthreads();
  for (int idx = tid; idx < 8192; idx += 256) {
    int r = idx >> 7, jc = idx & 127;
    float h1 = (float)Hh[r * 256 + jc] + mb11[jc];
    float h2 = (float)Hh[r * 256 + 128 + jc] + mb12[jc];
    float ge = 0.5f * h1 * (1.f + erff(h1 * 0.7071067811865475f));
    A[idx] = (_Float16)(ge * h2);
  }
  __syncthreads();
  const _Float16* W2 = w16 + OFF_M2;
  int lane = tid & 31, nloc = lane & 15, hi = lane >> 4;
  for (int tt = 0; tt < 4; ++tt) {
    int ti = wave * 4 + tt, mt = ti >> 3, nt = ti & 7;
    v8f c = {};
#pragma unroll
    for (int ks = 0; ks < 4; ++ks) {
      v16h a  = ldA(A + mt * 16 * 128 + ks * 32, 128);
      v16h bm = ldB(W2 + ks * 32 * 128 + nt * 16, 128);
      c = wmma(a, bm, c);
    }
    int d = nt * 16 + nloc;
#pragma unroll
    for (int r = 0; r < 8; ++r) {
      int row = mt * 16 + hi * 8 + r;
      size_t off = (size_t)d * HW + pix0 + row;
      out[off] = x1[off] + c[r] + mb2[d];
    }
  }
}

// ---------------- launch ----------------
extern "C" void kernel_launch(void* const* d_in, const int* in_sizes, int n_in,
                              void* d_out, int out_size, void* d_ws, size_t ws_size,
                              hipStream_t stream) {
  const float* x     = (const float*)d_in[0];
  const float* w_ea  = (const float*)d_in[1];
  const float* w_sa  = (const float*)d_in[2];
  const float* ln1_g = (const float*)d_in[3];
  const float* ln1_b = (const float*)d_in[4];
  const float* ln2_g = (const float*)d_in[5];
  const float* ln2_b = (const float*)d_in[6];
  const float* rpb   = (const float*)d_in[7];
  const float* fc11w = (const float*)d_in[8];
  const float* fc11b = (const float*)d_in[9];
  const float* fc12w = (const float*)d_in[10];
  const float* fc12b = (const float*)d_in[11];
  const float* fc2w  = (const float*)d_in[12];
  const float* fc2b  = (const float*)d_in[13];
  const float* m11w  = (const float*)d_in[14];
  const float* m11b  = (const float*)d_in[15];
  const float* m12w  = (const float*)d_in[16];
  const float* m12b  = (const float*)d_in[17];
  const float* m2w   = (const float*)d_in[18];
  const float* m2b   = (const float*)d_in[19];
  (void)in_sizes; (void)n_in; (void)out_size; (void)ws_size;

  char* ws = (char*)d_ws;
  size_t off = 0;
  auto carve = [&](size_t bytes) -> void* {
    void* p = ws + off;
    off += (bytes + 255) & ~(size_t)255;
    return p;
  };
  _Float16* xw_sa = (_Float16*)carve(2304ull * 2048 * 2);      // 9.4 MB
  _Float16* xw_ea = (_Float16*)carve(2304ull * 2048 * 2);      // 9.4 MB
  _Float16* w16   = (_Float16*)carve(W16_TOT * 2);             // 1 MB
  _Float16* y_ea  = (_Float16*)carve(2304ull * 64 * 384 * 2);  // 113 MB
  _Float16* y_sa  = (_Float16*)carve(2304ull * 64 * 384 * 2);  // 113 MB
  _Float16* feat  = (_Float16*)carve(576ull * 64 * 1536 * 2);  // 113 MB
  float*    x1    = (float*)carve(4718592ull * 4);             // 18.9 MB

  auto cvt = [&](const float* s, _Float16* d, int n) {
    k_cvt<<<(n + 255) / 256, 256, 0, stream>>>(s, d, n);
  };
  cvt(w_ea,  w16 + OFF_WEA,  12288);
  cvt(w_sa,  w16 + OFF_WSA,  12288);
  cvt(fc11w, w16 + OFF_FC11, 196608);
  cvt(fc12w, w16 + OFF_FC12, 196608);
  cvt(fc2w,  w16 + OFF_FC2,  16384);
  cvt(m11w,  w16 + OFF_M11,  16384);
  cvt(m12w,  w16 + OFF_M12,  16384);
  cvt(m2w,   w16 + OFF_M2,   16384);

  k_prep<<<2304, 64, 0, stream>>>(x, ln1_g, ln1_b, xw_sa, xw_ea);
  k_qkv<<<dim3(2304, 48), 32, 0, stream>>>(xw_ea, xw_sa, w16, y_ea, y_sa);
  k_ea<<<dim3(576, 4, 8), 32, 0, stream>>>(y_ea, feat);
  k_sa<<<dim3(576, 4), 128, 0, stream>>>(y_sa, rpb, feat);
  k_geglu1<<<576, 256, 0, stream>>>(feat, w16, fc11b, fc12b, fc2b, x, x1);
  k_mlp2<<<576, 256, 0, stream>>>(x1, ln2_g, ln2_b, w16, m11b, m12b, m2b,
                                  (float*)d_out);
}
